// HeteroNet_65163243815764
// MI455X (gfx1250) — matrix operand order repaired and checked by reading, not compiled, not run on hardware
//
#include <hip/hip_runtime.h>

typedef __attribute__((ext_vector_type(16))) _Float16 v16h;
typedef __attribute__((ext_vector_type(8)))  float    v8f;

constexpr int Tn   = 100000;   // tasks   (6250 * 16)
constexpr int Rn   = 40000;
constexpr int Dn   = 8000;
constexpr int Ln   = 150000;   // links   (9375 * 16)
constexpr int E_UR = 800000;
constexpr int E_UD = 200000;
constexpr int E_UL = 1600000;
constexpr int E_CT = 1500000;
constexpr int Hn   = 64;

// CDNA5 hardware V_TANH_F32 (single TRANS32 op) when available.
static __device__ __forceinline__ float dev_tanh(float x) {
#if __has_builtin(__builtin_amdgcn_tanhf)
  return __builtin_amdgcn_tanhf(x);
#else
  return tanhf(x);
#endif
}

// Order-preserving float <-> uint map so atomicMax/Min(u32) do float max/min.
static __device__ __forceinline__ unsigned enc_f32(float f) {
  unsigned u = __float_as_uint(f);
  return (u & 0x80000000u) ? ~u : (u | 0x80000000u);
}
static __device__ __forceinline__ float dec_f32(unsigned k) {
  unsigned u = (k & 0x80000000u) ? (k ^ 0x80000000u) : ~k;
  return __uint_as_float(u);
}

// ---- scalar segment sum of edge features into module nodes ----
__global__ void seg_sum_scalar(const float* __restrict__ feat, const int* __restrict__ dst,
                               float* __restrict__ s, int E) {
  int e = blockIdx.x * blockDim.x + threadIdx.x;
  if (e < E) atomicAdd(&s[dst[e]], feat[e]);
}

// ---- per-task scalar max/min + degree over reverse edges ----
__global__ void reduce_task(const float* __restrict__ s, const int* __restrict__ src,
                            const int* __restrict__ dst, unsigned* __restrict__ maxk,
                            unsigned* __restrict__ mink, int* __restrict__ cnt, int E) {
  int e = blockIdx.x * blockDim.x + threadIdx.x;
  if (e < E) {
    unsigned k = enc_f32(s[dst[e]]);
    int t = src[e];
    atomicMax(&maxk[t], k);
    atomicMin(&mink[t], k);
    atomicAdd(&cnt[t], 1);
  }
}

// ---- pre-pack a KxHn (row-major f32) weight matrix into WMMA B fragments ----
// Fragment layout (16-bit B, 32x16, wave32): lane = ksel*16 + n, half i <-> K = ksel*8 + i (i<8), 16 + ksel*8 + (i-8) (i>=8).
__global__ void prepack_B(const float* __restrict__ W, _Float16* __restrict__ out, int ktiles) {
  int tid = blockIdx.x * blockDim.x + threadIdx.x;
  if (tid >= ktiles * 128) return;
  int lane = tid & 31;
  int nt   = (tid >> 5) & 3;
  int kt   = tid >> 7;
  int col  = nt * 16 + (lane & 15);
  int ksel = lane >> 4;
  _Float16* o = out + (size_t)tid * 16;
  for (int i = 0; i < 16; ++i) {
    int k = kt * 32 + (i < 8 ? ksel * 8 + i : 16 + ksel * 8 + (i - 8));
    o[i] = (_Float16)W[k * Hn + col];
  }
}

// ---- task_h = tanh([h_r|h_d|h_l] @ Wt + bt) : 100000x192 @ 192x64 via WMMA ----
// A fragments synthesized in-register from scalar smax/smin (tanh monotonicity trick).
// Weight/bias loads are unconditional float4 vectors; the zero-degree mask is one cndmask.
__global__ __launch_bounds__(256)
void gemm_task(const unsigned* __restrict__ mxr, const unsigned* __restrict__ mnr, const int* __restrict__ cr,
               const unsigned* __restrict__ mxd, const unsigned* __restrict__ mnd, const int* __restrict__ cd,
               const unsigned* __restrict__ mxl, const unsigned* __restrict__ mnl, const int* __restrict__ cl,
               const float* __restrict__ Wr, const float* __restrict__ br,
               const float* __restrict__ Wd, const float* __restrict__ bd,
               const float* __restrict__ Wl, const float* __restrict__ bl,
               const _Float16* __restrict__ Bp, const float* __restrict__ bt,
               float* __restrict__ task_h) {
  int wave = threadIdx.x >> 5, lane = threadIdx.x & 31;
  int mtile = blockIdx.x * 8 + wave;
  if (mtile >= Tn / 16) return;                 // wave-uniform exit keeps EXEC all-ones
  int r = lane & 15, sel = lane >> 4;
  int t = mtile * 16 + r;

  const unsigned* MX[3] = {mxr, mxd, mxl};
  const unsigned* MN[3] = {mnr, mnd, mnl};
  const int*      CN[3] = {cr, cd, cl};
  const float*    WW[3] = {Wr, Wd, Wl};
  const float*    BB[3] = {br, bd, bl};

  v8f acc[4] = {};
  #pragma unroll
  for (int kt = 0; kt < 6; ++kt) {
    int rel = kt >> 1;                          // kt 0,1 -> ret; 2,3 -> dram; 4,5 -> link
    bool present = CN[rel][t] > 0;
    float smax = dec_f32(MX[rel][t]);
    float smin = dec_f32(MN[rel][t]);
    // Per lane the 16 needed columns are two contiguous 8-float runs.
    int h0 = ((kt & 1) << 5) + sel * 8;
    const float* Wp = WW[rel] + h0;
    const float* Vp = BB[rel] + h0;
    float4 w0 = *(const float4*)(Wp);
    float4 w1 = *(const float4*)(Wp + 4);
    float4 w2 = *(const float4*)(Wp + 16);
    float4 w3 = *(const float4*)(Wp + 20);
    float4 c0 = *(const float4*)(Vp);
    float4 c1 = *(const float4*)(Vp + 4);
    float4 c2 = *(const float4*)(Vp + 16);
    float4 c3 = *(const float4*)(Vp + 20);
    float wv[16] = {w0.x, w0.y, w0.z, w0.w, w1.x, w1.y, w1.z, w1.w,
                    w2.x, w2.y, w2.z, w2.w, w3.x, w3.y, w3.z, w3.w};
    float bv[16] = {c0.x, c0.y, c0.z, c0.w, c1.x, c1.y, c1.z, c1.w,
                    c2.x, c2.y, c2.z, c2.w, c3.x, c3.y, c3.z, c3.w};
    v16h a;
    #pragma unroll
    for (int i = 0; i < 16; ++i) {
      float w = wv[i];
      float tv = dev_tanh(w * (w > 0.0f ? smax : smin) + bv[i]);
      a[i] = (_Float16)(present ? tv : 0.0f);   // select, not branch
    }
    const _Float16* bp = Bp + (size_t)(kt * 4) * 512 + lane * 16;
    #pragma unroll
    for (int nt = 0; nt < 4; ++nt) {
      v16h b = *(const v16h*)(bp + nt * 512);
      acc[nt] = __builtin_amdgcn_wmma_f32_16x16x32_f16(false, a, false, b,
                                                       (short)0, acc[nt], false, false);
    }
  }
  // C/D layout: N = lane&15, M = sel*8 + vgpr
  #pragma unroll
  for (int nt = 0; nt < 4; ++nt)
    #pragma unroll
    for (int v = 0; v < 8; ++v) {
      int row = mtile * 16 + sel * 8 + v;
      int col = nt * 16 + r;
      task_h[(size_t)row * Hn + col] = dev_tanh(acc[nt][v] + bt[col]);
    }
}

__global__ void count_edges(const int* __restrict__ dst, int* __restrict__ cnt, int E) {
  int e = blockIdx.x * blockDim.x + threadIdx.x;
  if (e < E) atomicAdd(&cnt[dst[e]], 1);
}

// link_acc[l,:] += task_h[t,:] over use_link edges (one thread per edge,feature)
__global__ void mean_scatter(const float* __restrict__ task_h, const int* __restrict__ src,
                             const int* __restrict__ dst, float* __restrict__ acc, long long total) {
  long long idx = (long long)blockIdx.x * blockDim.x + threadIdx.x;
  if (idx >= total) return;
  int e = (int)(idx >> 6), h = (int)(idx & 63);
  atomicAdd(&acc[(size_t)dst[e] * Hn + h], task_h[(size_t)src[e] * Hn + h]);
}

__global__ void mean_div(float* __restrict__ acc, const int* __restrict__ cnt, int n) {
  int idx = blockIdx.x * blockDim.x + threadIdx.x;
  if (idx < n) acc[idx] /= fmaxf((float)cnt[idx >> 6], 1.0f);
}

// h_fwd / h_bwd accumulation over connect_to edges (both directions in one pass)
__global__ void ct_scatter(const float* __restrict__ fe, const float* __restrict__ be,
                           const int* __restrict__ src, const int* __restrict__ dst,
                           const float* __restrict__ linkh,
                           float* __restrict__ hf, float* __restrict__ hb, long long total) {
  long long idx = (long long)blockIdx.x * blockDim.x + threadIdx.x;
  if (idx >= total) return;
  int e = (int)(idx >> 6), h = (int)(idx & 63);
  int s = src[e], d = dst[e];
  atomicAdd(&hf[(size_t)d * Hn + h], fe[e] * linkh[(size_t)s * Hn + h]);
  atomicAdd(&hb[(size_t)s * Hn + h], be[e] * linkh[(size_t)d * Hn + h]);
}

// link_h += relu([hf|hb] @ W + b) : 150000x128 @ 128x64 via WMMA, residual in-place
__global__ __launch_bounds__(256)
void gemm_link(const float* __restrict__ hf, const float* __restrict__ hb,
               const _Float16* __restrict__ Bp, const float* __restrict__ bvec,
               float* __restrict__ linkh) {
  int wave = threadIdx.x >> 5, lane = threadIdx.x & 31;
  int mtile = blockIdx.x * 8 + wave;
  if (mtile >= Ln / 16) return;
  int r = lane & 15, sel = lane >> 4;

  v8f acc[4] = {};
  #pragma unroll
  for (int kt = 0; kt < 4; ++kt) {
    const float* Asrc = (kt < 2) ? hf : hb;     // cols 0..63 = h_fwd, 64..127 = h_bwd
    int kk0 = (kt & 1) * 32;
    const float* arow = Asrc + (size_t)(mtile * 16 + r) * Hn + kk0;
    // 16B-aligned vector loads (global_load_b128) then cvt to f16 fragment
    float4 f0 = *(const float4*)(arow + sel * 8);
    float4 f1 = *(const float4*)(arow + sel * 8 + 4);
    float4 f2 = *(const float4*)(arow + 16 + sel * 8);
    float4 f3 = *(const float4*)(arow + 16 + sel * 8 + 4);
    v16h a;
    a[0]  = (_Float16)f0.x; a[1]  = (_Float16)f0.y; a[2]  = (_Float16)f0.z; a[3]  = (_Float16)f0.w;
    a[4]  = (_Float16)f1.x; a[5]  = (_Float16)f1.y; a[6]  = (_Float16)f1.z; a[7]  = (_Float16)f1.w;
    a[8]  = (_Float16)f2.x; a[9]  = (_Float16)f2.y; a[10] = (_Float16)f2.z; a[11] = (_Float16)f2.w;
    a[12] = (_Float16)f3.x; a[13] = (_Float16)f3.y; a[14] = (_Float16)f3.z; a[15] = (_Float16)f3.w;
    const _Float16* bp = Bp + (size_t)(kt * 4) * 512 + lane * 16;
    #pragma unroll
    for (int nt = 0; nt < 4; ++nt) {
      v16h b = *(const v16h*)(bp + nt * 512);
      acc[nt] = __builtin_amdgcn_wmma_f32_16x16x32_f16(false, a, false, b,
                                                       (short)0, acc[nt], false, false);
    }
  }
  #pragma unroll
  for (int nt = 0; nt < 4; ++nt)
    #pragma unroll
    for (int v = 0; v < 8; ++v) {
      int row = mtile * 16 + sel * 8 + v;
      int col = nt * 16 + r;
      float x = fmaxf(acc[nt][v] + bvec[col], 0.0f);
      linkh[(size_t)row * Hn + col] += x;       // row-tile owned by this wave: no race
    }
}

extern "C" void kernel_launch(void* const* d_in, const int* in_sizes, int n_in,
                              void* d_out, int out_size, void* d_ws, size_t ws_size,
                              hipStream_t stream) {
  (void)in_sizes; (void)n_in; (void)out_size; (void)ws_size;
  const float* ur_feat = (const float*)d_in[0];
  const float* ud_feat = (const float*)d_in[1];
  const float* ul_feat = (const float*)d_in[2];
  const float* ct_feat = (const float*)d_in[3];
  const float* cb_feat = (const float*)d_in[4];
  const float* Wr = (const float*)d_in[5];
  const float* br = (const float*)d_in[6];
  const float* Wd = (const float*)d_in[7];
  const float* bd = (const float*)d_in[8];
  const float* Wl = (const float*)d_in[9];
  const float* bl = (const float*)d_in[10];
  const float* Wt = (const float*)d_in[11];
  const float* bt = (const float*)d_in[12];
  const float* W0 = (const float*)d_in[13];
  const float* b0 = (const float*)d_in[14];
  const float* W1 = (const float*)d_in[15];
  const float* b1 = (const float*)d_in[16];
  const int* ur_src = (const int*)d_in[17];
  const int* ur_dst = (const int*)d_in[18];
  const int* ud_src = (const int*)d_in[19];
  const int* ud_dst = (const int*)d_in[20];
  const int* ul_src = (const int*)d_in[21];
  const int* ul_dst = (const int*)d_in[22];
  const int* ct_src = (const int*)d_in[23];
  const int* ct_dst = (const int*)d_in[24];

  float* out    = (float*)d_out;
  float* link_h = out;                       // [L, 64]  (output 0, also working buffer)
  float* task_h = out + (size_t)Ln * Hn;     // [T, 64]  (output 1)

  // 256B-aligned bump allocator over workspace
  char* ws = (char*)d_ws;
  size_t off = 0;
  auto alloc = [&](size_t bytes) -> void* {
    void* p = ws + off;
    off = (off + bytes + 255) & ~(size_t)255;
    return p;
  };
  float*    s_r = (float*)alloc((size_t)Rn * 4);
  float*    s_d = (float*)alloc((size_t)Dn * 4);
  float*    s_l = (float*)alloc((size_t)Ln * 4);
  unsigned* mx[3]; unsigned* mn[3]; int* cn[3];
  for (int i = 0; i < 3; ++i) {
    mx[i] = (unsigned*)alloc((size_t)Tn * 4);
    mn[i] = (unsigned*)alloc((size_t)Tn * 4);
    cn[i] = (int*)alloc((size_t)Tn * 4);
  }
  _Float16* Btp  = (_Float16*)alloc((size_t)6 * 4 * 512 * 2);
  _Float16* B0p  = (_Float16*)alloc((size_t)4 * 4 * 512 * 2);
  _Float16* B1p  = (_Float16*)alloc((size_t)4 * 4 * 512 * 2);
  int*      cntl = (int*)alloc((size_t)Ln * 4);
  float*    hf   = (float*)alloc((size_t)Ln * Hn * 4);
  float*    hb   = (float*)alloc((size_t)Ln * Hn * 4);

  auto cdiv = [](long long a, long long b) { return (int)((a + b - 1) / b); };

  // --- per-launch init (graph-capture-safe memset nodes) ---
  hipMemsetAsync(s_r, 0, (size_t)Rn * 4, stream);
  hipMemsetAsync(s_d, 0, (size_t)Dn * 4, stream);
  hipMemsetAsync(s_l, 0, (size_t)Ln * 4, stream);
  for (int i = 0; i < 3; ++i) {
    hipMemsetAsync(mx[i], 0x00, (size_t)Tn * 4, stream);   // key 0 == -inf
    hipMemsetAsync(mn[i], 0xFF, (size_t)Tn * 4, stream);   // key ~0 == +inf
    hipMemsetAsync(cn[i], 0x00, (size_t)Tn * 4, stream);
  }
  hipMemsetAsync(link_h, 0, (size_t)Ln * Hn * 4, stream);
  hipMemsetAsync(cntl, 0, (size_t)Ln * 4, stream);

  // --- module scalar sums ---
  seg_sum_scalar<<<cdiv(E_UR, 256), 256, 0, stream>>>(ur_feat, ur_dst, s_r, E_UR);
  seg_sum_scalar<<<cdiv(E_UD, 256), 256, 0, stream>>>(ud_feat, ud_dst, s_d, E_UD);
  seg_sum_scalar<<<cdiv(E_UL, 256), 256, 0, stream>>>(ul_feat, ul_dst, s_l, E_UL);

  // --- reverse-edge scalar max/min per task ---
  reduce_task<<<cdiv(E_UR, 256), 256, 0, stream>>>(s_r, ur_src, ur_dst, mx[0], mn[0], cn[0], E_UR);
  reduce_task<<<cdiv(E_UD, 256), 256, 0, stream>>>(s_d, ud_src, ud_dst, mx[1], mn[1], cn[1], E_UD);
  reduce_task<<<cdiv(E_UL, 256), 256, 0, stream>>>(s_l, ul_src, ul_dst, mx[2], mn[2], cn[2], E_UL);

  // --- pre-pack weights into WMMA B fragments ---
  prepack_B<<<cdiv(6 * 128, 256), 256, 0, stream>>>(Wt, Btp, 6);
  prepack_B<<<cdiv(4 * 128, 256), 256, 0, stream>>>(W0, B0p, 4);
  prepack_B<<<cdiv(4 * 128, 256), 256, 0, stream>>>(W1, B1p, 4);

  // --- task_h GEMM (WMMA) ---
  gemm_task<<<cdiv(Tn / 16, 8), 256, 0, stream>>>(
      mx[0], mn[0], cn[0], mx[1], mn[1], cn[1], mx[2], mn[2], cn[2],
      Wr, br, Wd, bd, Wl, bl, Btp, bt, task_h);

  // --- link mean aggregation ---
  count_edges<<<cdiv(E_UL, 256), 256, 0, stream>>>(ul_dst, cntl, E_UL);
  mean_scatter<<<cdiv((long long)E_UL * Hn, 256), 256, 0, stream>>>(
      task_h, ul_src, ul_dst, link_h, (long long)E_UL * Hn);
  mean_div<<<cdiv((long long)Ln * Hn, 256), 256, 0, stream>>>(link_h, cntl, Ln * Hn);

  // --- 2 LinkConv layers ---
  const _Float16* Bps[2] = {B0p, B1p};
  const float*    bs[2]  = {b0, b1};
  for (int layer = 0; layer < 2; ++layer) {
    hipMemsetAsync(hf, 0, (size_t)Ln * Hn * 4, stream);
    hipMemsetAsync(hb, 0, (size_t)Ln * Hn * 4, stream);
    ct_scatter<<<cdiv((long long)E_CT * Hn, 256), 256, 0, stream>>>(
        ct_feat, cb_feat, ct_src, ct_dst, link_h, hf, hb, (long long)E_CT * Hn);
    gemm_link<<<cdiv(Ln / 16, 8), 256, 0, stream>>>(hf, hb, Bps[layer], bs[layer], link_h);
  }
}
